// ProteinGraph_72318659330279
// MI455X (gfx1250) — matrix operand order repaired and checked by reading, not compiled, not run on hardware
//
#include <hip/hip_runtime.h>
#include <hip/hip_bf16.h>
#include <math.h>

#define HID 64
#define IN_DIM 38
#define NUM_LAYERS 4
#define GMAX 2048

typedef __attribute__((ext_vector_type(16))) _Float16 v16h;
typedef __attribute__((ext_vector_type(8)))  float    v8f;

// ---------------------------------------------------------------- helpers
__device__ __forceinline__ void atomAddF(float* p, float v) {
  __hip_atomic_fetch_add(p, v, __ATOMIC_RELAXED, __HIP_MEMORY_SCOPE_AGENT);
}
__device__ __forceinline__ float leaky(float x) { return x >= 0.f ? x : 0.01f * x; }

// order-preserving float->u32 encoding for atomicMax
__device__ __forceinline__ unsigned encF(float f) {
  unsigned b = __float_as_uint(f);
  return (b & 0x80000000u) ? ~b : (b | 0x80000000u);
}
__device__ __forceinline__ float decF(unsigned k) {
  return (k & 0x80000000u) ? __uint_as_float(k ^ 0x80000000u) : __uint_as_float(~k);
}

// A-matrix (16x32 f16) half-index -> K   (ISA 7.12.2 table)
__device__ __forceinline__ int k_of_a(int i, int hi) {
  return i + hi * 8 + (i >= 8 ? 8 : 0);
}

__device__ __forceinline__ v8f wmma_f16(v16h a, v16h b, v8f c) {
  return __builtin_amdgcn_wmma_f32_16x16x32_f16(false, a, false, b, (short)0, c,
                                                false, false);
}

// ---------------------------------------------------------------- fills
__global__ void k_fill(float* p, float v, long long n) {
  long long i = (long long)blockIdx.x * blockDim.x + threadIdx.x;
  long long s = (long long)gridDim.x * blockDim.x;
  for (; i < n; i += s) p[i] = v;
}

// ---------------------------------------------------------------- encoder GEMM (N x 38) @ (38 x 64) + bias  (K zero-padded to 64)
// 128 threads = 4 waves; each wave computes a 16x16 WMMA tile; results staged
// in LDS, then written with coalesced float4 stores.
__global__ void k_gemm_enc(const float* __restrict__ X, const float* __restrict__ W,
                           const float* __restrict__ bias, float* __restrict__ out,
                           int N) {
  __shared__ float tile[16][HID + 1];
  int lane = threadIdx.x & 31;
  int wave = threadIdx.x >> 5;
  int hi = lane >> 4, lo = lane & 15;
  int m0 = blockIdx.x * 16, n0 = wave * 16;
  int row = m0 + lo; if (row >= N) row = N - 1;
  const float* xr = X + (long long)row * IN_DIM;
  int col = n0 + lo;

  v16h a0, a1, b0, b1;
#pragma unroll
  for (int i = 0; i < 16; i++) {
    int k = k_of_a(i, hi);
    a0[i] = (_Float16)(k < IN_DIM ? xr[k] : 0.f);
    int k1 = k + 32;
    a1[i] = (_Float16)(k1 < IN_DIM ? xr[k1] : 0.f);
  }
#pragma unroll
  for (int i = 0; i < 16; i++) {
    int k = k_of_a(i, hi);
    b0[i] = (_Float16)(k < IN_DIM ? W[k * HID + col] : 0.f);
    int k1 = k + 32;
    b1[i] = (_Float16)(k1 < IN_DIM ? W[k1 * HID + col] : 0.f);
  }
  v8f c = {};
  c = wmma_f16(a0, b0, c);
  c = wmma_f16(a1, b1, c);
  float bv = bias[col];
#pragma unroll
  for (int r = 0; r < 8; r++) tile[hi * 8 + r][col] = c[r] + bv;
  __syncthreads();

  // coalesced write-out: thread -> (row = tid/8, 8 contiguous cols)
  int r2 = threadIdx.x >> 3;
  int c2 = (threadIdx.x & 7) * 8;
  int gm = m0 + r2;
  if (gm < N) {
    float4 lo4 = make_float4(tile[r2][c2 + 0], tile[r2][c2 + 1],
                             tile[r2][c2 + 2], tile[r2][c2 + 3]);
    float4 hi4 = make_float4(tile[r2][c2 + 4], tile[r2][c2 + 5],
                             tile[r2][c2 + 6], tile[r2][c2 + 7]);
    float* dst = out + (long long)gm * HID + c2;
    *(float4*)(dst + 0) = lo4;
    *(float4*)(dst + 4) = hi4;
  }
}

// ---------------------------------------------------------------- batchnorm stats (sum / sumsq per column)
__global__ void k_bn_stats(const float* __restrict__ x, float* __restrict__ stats,
                           int N) {
  __shared__ float ls[128];
  if (threadIdx.x < 128) ls[threadIdx.x] = 0.f;
  __syncthreads();
  int col = threadIdx.x & 63;
  long long total = (long long)N * HID;
  long long stride = (long long)gridDim.x * blockDim.x;  // multiple of 64
  float s = 0.f, q = 0.f;
  for (long long i = (long long)blockIdx.x * blockDim.x + threadIdx.x; i < total;
       i += stride) {
    float xv = x[i];
    s += xv; q += xv * xv;
  }
  atomicAdd(&ls[col], s);
  atomicAdd(&ls[64 + col], q);
  __syncthreads();
  if (threadIdx.x < 128) atomAddF(&stats[threadIdx.x], ls[threadIdx.x]);
}

__global__ void k_bn_finalize(const float* __restrict__ stats,
                              const float* __restrict__ g, const float* __restrict__ b,
                              float* __restrict__ scale, float* __restrict__ shift,
                              int N) {
  int c = threadIdx.x;
  if (c >= HID) return;
  float m = stats[c] / (float)N;
  float v = stats[64 + c] / (float)N - m * m;
  float sc = g[c] * rsqrtf(v + 1e-5f);
  scale[c] = sc;
  shift[c] = b[c] - m * sc;
}

// feat = x*scale+shift ; acc = feat
__global__ void k_bn_apply(const float* __restrict__ x, const float* __restrict__ scale,
                           const float* __restrict__ shift, float* __restrict__ feat,
                           float* __restrict__ acc, int N) {
  long long total = (long long)N * HID;
  long long stride = (long long)gridDim.x * blockDim.x;
  for (long long i = (long long)blockIdx.x * blockDim.x + threadIdx.x; i < total;
       i += stride) {
    int col = (int)(i & 63);
    float y = x[i] * scale[col] + shift[col];
    feat[i] = y;
    acc[i] = y;
  }
}

// ---------------------------------------------------------------- per-node attention scores s_u, s_v
__global__ void k_node_scores(const float* __restrict__ feat,
                              const float* __restrict__ aw,  // [128]
                              float* __restrict__ su, float* __restrict__ sv, int N) {
  __shared__ float saw[128];
  if (threadIdx.x < 128) saw[threadIdx.x] = aw[threadIdx.x];
  __syncthreads();
  int n = blockIdx.x * blockDim.x + threadIdx.x;
  if (n >= N) return;
  const float* f = feat + (long long)n * HID;
  float a = 0.f, b = 0.f;
#pragma unroll
  for (int h = 0; h < HID; h++) {
    float x = f[h];
    a += x * saw[h];
    b += x * saw[HID + h];
  }
  su[n] = a; sv[n] = b;
}

// ---------------------------------------------------------------- edge pass A: attn + segment max
__global__ void k_edge_attn(const int* __restrict__ u, const int* __restrict__ v,
                            const float* __restrict__ su, const float* __restrict__ sv,
                            float* __restrict__ attn, unsigned* __restrict__ amax,
                            int E) {
  int e = blockIdx.x * blockDim.x + threadIdx.x;
  if (e >= E) return;
  float a = su[u[e]] + sv[v[e]];
  attn[e] = a;
  atomicMax(&amax[v[e]], encF(a));
}

// ---------------------------------------------------------------- edge pass B: exp + segment sum
__global__ void k_edge_exp(const int* __restrict__ v, float* __restrict__ attn,
                           const unsigned* __restrict__ amax,
                           float* __restrict__ denom, int E) {
  int e = blockIdx.x * blockDim.x + threadIdx.x;
  if (e >= E) return;
  int vv = v[e];
  float ex = __expf(attn[e] - decF(amax[vv]));
  attn[e] = ex;
  atomAddF(&denom[vv], ex);
}

// ---------------------------------------------------------------- edge pass C: weighted message scatter (16 thr/edge, float4)
__global__ void k_edge_msg(const int* __restrict__ u, const int* __restrict__ v,
                           const float* __restrict__ attn,
                           const float* __restrict__ denom,
                           const float* __restrict__ feat, float* __restrict__ msg,
                           int E) {
  int t = blockIdx.x * blockDim.x + threadIdx.x;
  int e = t >> 4, c = t & 15;
  if (e >= E) return;
  if (e + 2048 < E) __builtin_prefetch(&attn[e + 2048], 0, 0);
  int uu = u[e], vv = v[e];
  float w = attn[e] / denom[vv];
  const float4 f = *(const float4*)(feat + (long long)uu * HID + c * 4);
  float* dst = msg + (long long)vv * HID + c * 4;
  atomAddF(dst + 0, f.x * w);
  atomAddF(dst + 1, f.y * w);
  atomAddF(dst + 2, f.z * w);
  atomAddF(dst + 3, f.w * w);
}

// ---------------------------------------------------------------- fused (msg @ lw + lb) -> LayerNorm -> leaky ; feat=y ; acc+=y
__global__ void k_gemm_ln(const float* __restrict__ msg, const float* __restrict__ lw,
                          const float* __restrict__ lb, const float* __restrict__ lng,
                          const float* __restrict__ lnb, float* __restrict__ feat,
                          float* __restrict__ acc, int N) {
  __shared__ float tile[16][HID + 1];
  __shared__ float smu[16], srs[16];
  int wave = threadIdx.x >> 5, lane = threadIdx.x & 31;
  int hi = lane >> 4, lo = lane & 15;
  int m0 = blockIdx.x * 16, n0 = wave * 16;
  int row = m0 + lo; if (row >= N) row = N - 1;
  const float* mr = msg + (long long)row * HID;
  int col = n0 + lo;

  v16h a0, a1, b0, b1;
#pragma unroll
  for (int i = 0; i < 16; i++) {
    int k = k_of_a(i, hi);
    a0[i] = (_Float16)mr[k];
    a1[i] = (_Float16)mr[k + 32];
  }
#pragma unroll
  for (int i = 0; i < 16; i++) {
    int k = k_of_a(i, hi);
    b0[i] = (_Float16)lw[k * HID + col];
    b1[i] = (_Float16)lw[(k + 32) * HID + col];
  }
  v8f c = {};
  c = wmma_f16(a0, b0, c);
  c = wmma_f16(a1, b1, c);
  float bv = lb[col];
#pragma unroll
  for (int r = 0; r < 8; r++) tile[hi * 8 + r][col] = c[r] + bv;
  __syncthreads();

  if (threadIdx.x < 16) {
    float s = 0.f, q = 0.f;
#pragma unroll
    for (int h = 0; h < HID; h++) { float x = tile[threadIdx.x][h]; s += x; q += x * x; }
    float m = s * (1.f / HID);
    float v = q * (1.f / HID) - m * m;
    smu[threadIdx.x] = m;
    srs[threadIdx.x] = rsqrtf(v + 1e-5f);
  }
  __syncthreads();

  // coalesced write-out: thread -> (row = tid/8, 8 contiguous cols)
  int r2 = threadIdx.x >> 3;
  int c2 = (threadIdx.x & 7) * 8;
  int gm = m0 + r2;
  if (gm < N) {
    float m = smu[r2], rs = srs[r2];
    float y[8];
#pragma unroll
    for (int j = 0; j < 8; j++) {
      int h = c2 + j;
      y[j] = leaky((tile[r2][h] - m) * rs * lng[h] + lnb[h]);
    }
    float* fdst = feat + (long long)gm * HID + c2;
    float* adst = acc + (long long)gm * HID + c2;
    float4 y0 = make_float4(y[0], y[1], y[2], y[3]);
    float4 y1 = make_float4(y[4], y[5], y[6], y[7]);
    float4 a0v = *(const float4*)(adst + 0);
    float4 a1v = *(const float4*)(adst + 4);
    *(float4*)(fdst + 0) = y0;
    *(float4*)(fdst + 4) = y1;
    *(float4*)(adst + 0) = make_float4(a0v.x + y0.x, a0v.y + y0.y,
                                       a0v.z + y0.z, a0v.w + y0.w);
    *(float4*)(adst + 4) = make_float4(a1v.x + y1.x, a1v.y + y1.y,
                                       a1v.z + y1.z, a1v.w + y1.w);
  }
}

// ---------------------------------------------------------------- graph pooling scatter (16 thr/node)
__global__ void k_scatter_graph(const float* __restrict__ acc,
                                const int* __restrict__ batch,
                                float* __restrict__ gsums, float* __restrict__ gcnt,
                                int N) {
  int t = blockIdx.x * blockDim.x + threadIdx.x;
  int n = t >> 4, c = t & 15;
  if (n >= N) return;
  int g = batch[n];
  const float inv = 1.f / (NUM_LAYERS + 1);
  const float4 f = *(const float4*)(acc + (long long)n * HID + c * 4);
  float* dst = gsums + (long long)g * HID + c * 4;
  atomAddF(dst + 0, f.x * inv);
  atomAddF(dst + 1, f.y * inv);
  atomAddF(dst + 2, f.z * inv);
  atomAddF(dst + 3, f.w * inv);
  if (c == 0) atomAddF(&gcnt[g], 1.f);
}

// ---------------------------------------------------------------- final: mean / BN / leaky  (single block, 64 threads)
__global__ void k_final(const float* __restrict__ gsums, const float* __restrict__ gcnt,
                        const int* __restrict__ ngp, const float* __restrict__ g2,
                        const float* __restrict__ b2, float* __restrict__ out) {
  int c = threadIdx.x;
  if (c >= HID) return;
  int ng = ngp[0];
  float s = 0.f, q = 0.f;
  for (int g = 0; g < ng; g++) {
    float x = gsums[g * HID + c] / fmaxf(gcnt[g], 1.f);
    s += x; q += x * x;
  }
  float m = s / (float)ng;
  float v = q / (float)ng - m * m;
  float rs = rsqrtf(v + 1e-5f);
  float gg = g2[c], bb = b2[c];
  for (int g = 0; g < ng; g++) {
    float x = gsums[g * HID + c] / fmaxf(gcnt[g], 1.f);
    out[g * HID + c] = leaky((x - m) * rs * gg + bb);
  }
}

// ================================================================ launch
extern "C" void kernel_launch(void* const* d_in, const int* in_sizes, int n_in,
                              void* d_out, int out_size, void* d_ws, size_t ws_size,
                              hipStream_t stream) {
  const int E = in_sizes[0] / 2;
  const int N = in_sizes[2];

  const int*   edge  = (const int*)d_in[0];
  const int*   u     = edge;
  const int*   v     = edge + E;
  const float* nfeat = (const float*)d_in[1];
  const int*   batch = (const int*)d_in[2];
  const int*   ngp   = (const int*)d_in[3];
  const float* Wenc  = (const float*)d_in[4];
  const float* benc  = (const float*)d_in[5];
  const float* bn1g  = (const float*)d_in[6];
  const float* bn1b  = (const float*)d_in[7];
  const float* a_w   = (const float*)d_in[8];
  const float* lin_w = (const float*)d_in[9];
  const float* lin_b = (const float*)d_in[10];
  const float* ln_g  = (const float*)d_in[11];
  const float* ln_b  = (const float*)d_in[12];
  const float* bn2g  = (const float*)d_in[13];
  const float* bn2b  = (const float*)d_in[14];
  float* out = (float*)d_out;

  // ---- workspace carving
  char* ws = (char*)d_ws;
  size_t off = 0;
  auto carve = [&](size_t bytes) -> char* {
    char* p = ws + off;
    off = (off + bytes + 255) & ~(size_t)255;
    return p;
  };
  float*    feat   = (float*)carve((size_t)N * HID * 4);
  float*    acc    = (float*)carve((size_t)N * HID * 4);
  float*    msg    = (float*)carve((size_t)N * HID * 4);
  float*    attn   = (float*)carve((size_t)E * 4);
  float*    su     = (float*)carve((size_t)N * 4);
  float*    sv     = (float*)carve((size_t)N * 4);
  unsigned* amax   = (unsigned*)carve((size_t)N * 4);
  float*    denom  = (float*)carve((size_t)N * 4);
  float*    stats  = (float*)carve(128 * 4);
  float*    scale  = (float*)carve(HID * 4);
  float*    shift  = (float*)carve(HID * 4);
  float*    gsums  = (float*)carve((size_t)GMAX * HID * 4);
  float*    gcnt   = (float*)carve((size_t)GMAX * 4);
  (void)ws_size;

  const int TB = 256;
  const int gemmBlocks = (N + 15) / 16;
  const int nodeBlocks = (N + TB - 1) / TB;
  const int edgeBlocks = (E + TB - 1) / TB;
  const int edge16Blocks = (int)(((long long)E * 16 + TB - 1) / TB);
  const int node16Blocks = (int)(((long long)N * 16 + TB - 1) / TB);
  const int elemBlocks = 2048;

  // ---- encoder: Z = X@W + b  (into msg as temp), then BN -> feat, acc
  k_gemm_enc<<<gemmBlocks, 128, 0, stream>>>(nfeat, Wenc, benc, msg, N);
  k_fill<<<1, 128, 0, stream>>>(stats, 0.f, 128);
  k_bn_stats<<<1024, TB, 0, stream>>>(msg, stats, N);
  k_bn_finalize<<<1, 64, 0, stream>>>(stats, bn1g, bn1b, scale, shift, N);
  k_bn_apply<<<elemBlocks, TB, 0, stream>>>(msg, scale, shift, feat, acc, N);

  // ---- GAT layers
  for (int l = 0; l < NUM_LAYERS; l++) {
    k_fill<<<elemBlocks, TB, 0, stream>>>(msg, 0.f, (long long)N * HID);
    k_fill<<<nodeBlocks, TB, 0, stream>>>((float*)amax, 0.f, N);   // encF identity
    k_fill<<<nodeBlocks, TB, 0, stream>>>(denom, 0.f, N);

    k_node_scores<<<nodeBlocks, TB, 0, stream>>>(feat, a_w + l * 2 * HID, su, sv, N);
    k_edge_attn<<<edgeBlocks, TB, 0, stream>>>(u, v, su, sv, attn, amax, E);
    k_edge_exp<<<edgeBlocks, TB, 0, stream>>>(v, attn, amax, denom, E);
    k_edge_msg<<<edge16Blocks, TB, 0, stream>>>(u, v, attn, denom, feat, msg, E);

    k_gemm_ln<<<gemmBlocks, 128, 0, stream>>>(msg, lin_w + l * HID * HID,
                                              lin_b + l * HID, ln_g + l * HID,
                                              ln_b + l * HID, feat, acc, N);
  }

  // ---- graph pooling + final BN + leaky
  k_fill<<<512, TB, 0, stream>>>(gsums, 0.f, (long long)GMAX * HID);
  k_fill<<<8, TB, 0, stream>>>(gcnt, 0.f, GMAX);
  k_scatter_graph<<<node16Blocks, TB, 0, stream>>>(acc, batch, gsums, gcnt, N);
  k_final<<<1, 64, 0, stream>>>(gsums, gcnt, ngp, bn2g, bn2b, out);
}
